// InterpretableMultiHeadAttention_48687749267757
// MI455X (gfx1250) — matrix operand order repaired and checked by reading, not compiled, not run on hardware
//
#include <hip/hip_runtime.h>
#include <hip/hip_bf16.h>
#include <stdint.h>

// Problem constants (from the reference)
#define B_   4
#define T_   1024
#define HID_ 1024
#define NH_  16
#define DH_  64

typedef __attribute__((ext_vector_type(16))) __bf16 v16bf;
typedef __attribute__((ext_vector_type(8)))  float  v8f;
typedef unsigned short u16;
typedef unsigned int   u32;

struct __attribute__((aligned(16))) U128 { u32 x, y, z, w; };
union Frag { U128 u[2]; v16bf v; };

__device__ __forceinline__ u16 f2bf(float f) {
  u32 u = __float_as_uint(f);
  u32 r = u + 0x7FFFu + ((u >> 16) & 1u);   // round-to-nearest-even
  return (u16)(r >> 16);
}
__device__ __forceinline__ float bf2f(u16 h) {
  return __uint_as_float(((u32)h) << 16);
}

// ---------------- conversion / transpose helpers ----------------

__global__ void k_f32_to_bf16(const float* __restrict__ in, u16* __restrict__ out, int n) {
  int i = blockIdx.x * blockDim.x + threadIdx.x;
  if (i < n) out[i] = f2bf(in[i]);
}

// out[n*K + k] = bf16(in[k*N + n])   (weights -> [N][K] for WMMA B-fragments)
__global__ void k_transpose_f32_to_bf16(const float* __restrict__ in, u16* __restrict__ out,
                                        int K, int N) {
  int i = blockIdx.x * blockDim.x + threadIdx.x;
  if (i >= K * N) return;
  int n = i / K, k = i % K;
  out[i] = f2bf(in[(size_t)k * N + n]);
}

// Vt[(b*DH + d)*T + t] = V[(b*T + t)*DH + d]
__global__ void k_transpose_v(const u16* __restrict__ V, u16* __restrict__ Vt) {
  int i = blockIdx.x * blockDim.x + threadIdx.x;     // over B*DH*T
  if (i >= B_ * DH_ * T_) return;
  int b = i / (DH_ * T_);
  int rem = i % (DH_ * T_);
  int d = rem / T_, t = rem % T_;
  Vt[i] = V[((size_t)(b * T_ + t)) * DH_ + d];
}

// ---------------- generic bf16 WMMA GEMM: C[M,N] = A[M,K] @ Bt[N,K]^T + bias ----------------
// 8 waves / block; each wave computes a 32x64 tile (2 M-tiles x 4 N-tiles).
// All 6 fragments of a k-chunk are preloaded into distinct registers before the
// WMMA chain so loadcnt waits can be staged instead of serializing each WMMA.

__global__ void __launch_bounds__(256)
k_gemm_bf16(const u16* __restrict__ A, const u16* __restrict__ Bt,
            const float* __restrict__ bias, void* __restrict__ Cout,
            int M, int N, int Kd, int storeF32) {
  const int wave = threadIdx.x >> 5;
  const int lane = threadIdx.x & 31;
  const int r  = lane & 15;
  const int hi = lane >> 4;
  const int m0 = blockIdx.y * 256 + wave * 32;
  const int n0 = blockIdx.x * 64;
  if (m0 >= M) return;

  v8f acc[2][4] = {};
  const u16* Arow0 = A + (size_t)(m0 + r) * Kd;
  const u16* Arow1 = A + (size_t)(m0 + 16 + r) * Kd;

  for (int k0 = 0; k0 < Kd; k0 += 32) {
    if (k0 + 32 < Kd) {                      // global_prefetch_b8 next A chunk
      __builtin_prefetch(Arow0 + k0 + 32, 0, 0);
      __builtin_prefetch(Arow1 + k0 + 32, 0, 0);
    }
    Frag a0, a1, bF[4];
    a0.u[0] = *reinterpret_cast<const U128*>(Arow0 + k0 + hi * 8);
    a0.u[1] = *reinterpret_cast<const U128*>(Arow0 + k0 + hi * 8 + 16);
    a1.u[0] = *reinterpret_cast<const U128*>(Arow1 + k0 + hi * 8);
    a1.u[1] = *reinterpret_cast<const U128*>(Arow1 + k0 + hi * 8 + 16);
#pragma unroll
    for (int nt = 0; nt < 4; ++nt) {
      const u16* Brow = Bt + (size_t)(n0 + nt * 16 + r) * Kd + k0 + hi * 8;
      bF[nt].u[0] = *reinterpret_cast<const U128*>(Brow);
      bF[nt].u[1] = *reinterpret_cast<const U128*>(Brow + 16);
    }
#pragma unroll
    for (int nt = 0; nt < 4; ++nt)
      acc[0][nt] = __builtin_amdgcn_wmma_f32_16x16x32_bf16(
          false, a0.v, false, bF[nt].v, (short)0, acc[0][nt], false, false);
#pragma unroll
    for (int nt = 0; nt < 4; ++nt)
      acc[1][nt] = __builtin_amdgcn_wmma_f32_16x16x32_bf16(
          false, a1.v, false, bF[nt].v, (short)0, acc[1][nt], false, false);
  }

#pragma unroll
  for (int mt = 0; mt < 2; ++mt) {
#pragma unroll
    for (int nt = 0; nt < 4; ++nt) {
      int n = n0 + nt * 16 + r;
      float bb = bias ? bias[n] : 0.0f;
#pragma unroll
      for (int rr = 0; rr < 8; ++rr) {
        int m = m0 + mt * 16 + rr + 8 * hi;
        float v = acc[mt][nt][rr] + bb;
        if (storeF32) ((float*)Cout)[(size_t)m * N + n] = v;
        else          ((u16*)Cout)[(size_t)m * N + n]   = f2bf(v);
      }
    }
  }
}

// ---------------- causal attention: one wave per (b, h, 16-row q tile) ----------------
// scores (WMMA) -> bf16 LDS -> fp32 softmax -> bf16 P in LDS -> P@V (WMMA)

__global__ void __launch_bounds__(32)
k_attention(const u16* __restrict__ Q, const u16* __restrict__ K,
            const u16* __restrict__ Vt, float* __restrict__ attn_out) {
  __shared__ u16 Sh[16 * 1024];   // bf16 scores, row-major [16][T]
  __shared__ u16 Ph[16 * 1024];   // bf16 probabilities

  const int lane = threadIdx.x;
  const int r  = lane & 15;
  const int hi = lane >> 4;
  const int qt = blockIdx.x, h = blockIdx.y, b = blockIdx.z;

  // Q A-fragments (16 rows x 64 d, as two K=32 fragments) straight from global
  const u16* Qrow = Q + ((size_t)(b * T_ + qt * 16 + r)) * (NH_ * DH_) + h * DH_;
  Frag qa0, qa1;
  qa0.u[0] = *reinterpret_cast<const U128*>(Qrow + hi * 8);
  qa0.u[1] = *reinterpret_cast<const U128*>(Qrow + hi * 8 + 16);
  qa1.u[0] = *reinterpret_cast<const U128*>(Qrow + 32 + hi * 8);
  qa1.u[1] = *reinterpret_cast<const U128*>(Qrow + 32 + hi * 8 + 16);

  const float scale = 0.125f;  // DH^-0.5

  for (int kt = 0; kt <= qt; ++kt) {
    // B-fragment: column n = kpos (r), K-dim = d, contiguous in K's storage
    const u16* Krow = K + ((size_t)(b * T_ + kt * 16 + r)) * (NH_ * DH_) + h * DH_;
    Frag b0, b1;
    b0.u[0] = *reinterpret_cast<const U128*>(Krow + hi * 8);
    b0.u[1] = *reinterpret_cast<const U128*>(Krow + hi * 8 + 16);
    b1.u[0] = *reinterpret_cast<const U128*>(Krow + 32 + hi * 8);
    b1.u[1] = *reinterpret_cast<const U128*>(Krow + 32 + hi * 8 + 16);
    v8f s = {};
    s = __builtin_amdgcn_wmma_f32_16x16x32_bf16(false, qa0.v, false, b0.v, (short)0, s, false, false);
    s = __builtin_amdgcn_wmma_f32_16x16x32_bf16(false, qa1.v, false, b1.v, (short)0, s, false, false);
    int col = kt * 16 + r;
#pragma unroll
    for (int rr = 0; rr < 8; ++rr) {
      int m  = rr + 8 * hi;
      int qg = qt * 16 + m;
      float v = s[rr] * scale;
      if (col > qg) v = -1e30f;          // causal mask (only fires on diagonal tile)
      Sh[m * 1024 + col] = f2bf(v);
    }
  }
  __syncthreads();

  const int L  = (qt + 1) * 16;
  const int Lp = (L + 31) & ~31;         // pad to K=32 multiple for WMMA
  if (lane < 16) {
    const u16* srow = &Sh[lane * 1024];
    u16*       prow = &Ph[lane * 1024];
    float mx = -1e30f;
    for (int j = 0; j < L; ++j) mx = fmaxf(mx, bf2f(srow[j]));
    float sum = 0.0f;
    for (int j = 0; j < L; ++j) {
      float p = __expf(bf2f(srow[j]) - mx);
      sum += p;
      prow[j] = f2bf(p);
    }
    for (int j = L; j < Lp; ++j) prow[j] = 0;
    u32 inv = __float_as_uint(1.0f / sum);   // stash 1/rowsum in (now dead) Sh row head
    Sh[lane * 1024 + 0] = (u16)(inv & 0xFFFFu);
    Sh[lane * 1024 + 1] = (u16)(inv >> 16);
  }
  __syncthreads();

  // O[16 x 64] = P[16 x L] @ V[L x 64], streaming Vt (V transposed: [d][t]);
  // all 4 V fragments preloaded before the WMMA chain.
  v8f o[4] = {};
  for (int k0 = 0; k0 < Lp; k0 += 32) {
    Frag a, bF[4];
    a.u[0] = *reinterpret_cast<const U128*>(&Ph[r * 1024 + k0 + hi * 8]);
    a.u[1] = *reinterpret_cast<const U128*>(&Ph[r * 1024 + k0 + hi * 8 + 16]);
#pragma unroll
    for (int nt = 0; nt < 4; ++nt) {
      const u16* vb = Vt + ((size_t)(b * DH_ + nt * 16 + r)) * T_ + k0 + hi * 8;
      bF[nt].u[0] = *reinterpret_cast<const U128*>(vb);
      bF[nt].u[1] = *reinterpret_cast<const U128*>(vb + 16);
    }
#pragma unroll
    for (int nt = 0; nt < 4; ++nt)
      o[nt] = __builtin_amdgcn_wmma_f32_16x16x32_bf16(
          false, a.v, false, bF[nt].v, (short)0, o[nt], false, false);
  }

  float* outb = attn_out + (((size_t)(b * NH_ + h)) * T_ + qt * 16) * DH_;
#pragma unroll
  for (int rr = 0; rr < 8; ++rr) {
    int m = rr + 8 * hi;
    u32 lo = Sh[m * 1024 + 0], hw = Sh[m * 1024 + 1];
    float isv = __uint_as_float(lo | (hw << 16));
#pragma unroll
    for (int nt = 0; nt < 4; ++nt)
      outb[(size_t)m * DH_ + nt * 16 + r] = o[nt][rr] * isv;
  }
}

// ---------------- mean over heads, cast to bf16 for final GEMM ----------------

__global__ void k_mean_heads(const float* __restrict__ attn, u16* __restrict__ mb) {
  int i = blockIdx.x * blockDim.x + threadIdx.x;   // over B*T*DH
  if (i >= B_ * T_ * DH_) return;
  int b = i / (T_ * DH_);
  int rem = i % (T_ * DH_);
  int t = rem / DH_, d = rem % DH_;
  float s = 0.0f;
#pragma unroll
  for (int h = 0; h < NH_; ++h)
    s += attn[(((size_t)(b * NH_ + h)) * T_ + t) * DH_ + d];
  mb[i] = f2bf(s * (1.0f / NH_));
}

// ---------------- host-side orchestration ----------------

extern "C" void kernel_launch(void* const* d_in, const int* in_sizes, int n_in,
                              void* d_out, int out_size, void* d_ws, size_t ws_size,
                              hipStream_t stream) {
  (void)in_sizes; (void)n_in; (void)out_size; (void)ws_size;
  const float* x  = (const float*)d_in[0];
  const float* Wq = (const float*)d_in[1];
  const float* bq = (const float*)d_in[2];
  const float* Wk = (const float*)d_in[3];
  const float* bk = (const float*)d_in[4];
  const float* Wv = (const float*)d_in[5];
  const float* bv = (const float*)d_in[6];
  const float* Wo = (const float*)d_in[7];

  char* ws = (char*)d_ws;
  size_t off = 0;
  auto carve = [&](size_t bytes) -> char* {
    char* p = ws + off;
    off += (bytes + 255) & ~(size_t)255;
    return p;
  };
  u16* xbf  = (u16*)carve((size_t)B_ * T_ * HID_ * 2);   // 8 MB
  u16* WqT  = (u16*)carve((size_t)HID_ * HID_ * 2);      // 2 MB
  u16* WkT  = (u16*)carve((size_t)HID_ * HID_ * 2);      // 2 MB
  u16* WvT  = (u16*)carve((size_t)DH_ * HID_ * 2);       // 128 KB  [N=64][K=1024]
  u16* WoT  = (u16*)carve((size_t)HID_ * DH_ * 2);       // 128 KB  [N=1024][K=64]
  u16* Qbf  = (u16*)carve((size_t)B_ * T_ * HID_ * 2);   // 8 MB
  u16* Kbf  = (u16*)carve((size_t)B_ * T_ * HID_ * 2);   // 8 MB
  u16* Vbf  = (u16*)carve((size_t)B_ * T_ * DH_ * 2);    // 512 KB
  u16* Vtbf = (u16*)carve((size_t)B_ * DH_ * T_ * 2);    // 512 KB
  u16* Mbf  = (u16*)carve((size_t)B_ * T_ * DH_ * 2);    // 512 KB

  const int TPB = 256;
  int nX = B_ * T_ * HID_;
  k_f32_to_bf16<<<(nX + TPB - 1) / TPB, TPB, 0, stream>>>(x, xbf, nX);
  k_transpose_f32_to_bf16<<<(HID_ * HID_ + TPB - 1) / TPB, TPB, 0, stream>>>(Wq, WqT, HID_, HID_);
  k_transpose_f32_to_bf16<<<(HID_ * HID_ + TPB - 1) / TPB, TPB, 0, stream>>>(Wk, WkT, HID_, HID_);
  k_transpose_f32_to_bf16<<<(HID_ * DH_ + TPB - 1) / TPB, TPB, 0, stream>>>(Wv, WvT, HID_, DH_);
  k_transpose_f32_to_bf16<<<(DH_ * HID_ + TPB - 1) / TPB, TPB, 0, stream>>>(Wo, WoT, DH_, HID_);

  // 8 waves/block, each wave a 32x64 tile -> block covers 256 rows x 64 cols
  dim3 gqk(HID_ / 64, (B_ * T_) / 256);   // (16, 16)
  k_gemm_bf16<<<gqk, 256, 0, stream>>>(xbf, WqT, bq, Qbf, B_ * T_, HID_, HID_, 0);
  k_gemm_bf16<<<gqk, 256, 0, stream>>>(xbf, WkT, bk, Kbf, B_ * T_, HID_, HID_, 0);
  dim3 gv(DH_ / 64, (B_ * T_) / 256);     // (1, 16)
  k_gemm_bf16<<<gv, 256, 0, stream>>>(xbf, WvT, bv, Vbf, B_ * T_, DH_, HID_, 0);

  k_transpose_v<<<(B_ * DH_ * T_ + TPB - 1) / TPB, TPB, 0, stream>>>(Vbf, Vtbf);

  float* attn_out = (float*)d_out + (size_t)B_ * T_ * HID_;   // second output: [B,NH,T,DH]
  dim3 ga(T_ / 16, NH_, B_);
  k_attention<<<ga, 32, 0, stream>>>(Qbf, Kbf, Vtbf, attn_out);

  k_mean_heads<<<(B_ * T_ * DH_ + TPB - 1) / TPB, TPB, 0, stream>>>(attn_out, Mbf);

  dim3 go(HID_ / 64, (B_ * T_) / 256);
  k_gemm_bf16<<<go, 256, 0, stream>>>(Mbf, WoT, nullptr, d_out, B_ * T_, HID_, DH_, 1);
}
